// LanguageDecoder_72318659330746
// MI455X (gfx1250) — matrix operand order repaired and checked by reading, not compiled, not run on hardware
//
#include <hip/hip_runtime.h>
#include <hip/hip_bf16.h>

// ---------------- problem dimensions ----------------
#define BATCH   32
#define T_ENC   512
#define DHID    1024
#define MAX_T   128
#define VOCAB   32000
#define DEMB    512
#define KCAT    (DHID + DEMB + DHID)      // 2560 : [weighted | e_t | h]
#define NGATES  (4 * DHID)                // 4096
#define KCONT   (DHID + DHID + DEMB)      // 2560 : [h_t | weighted | e_t]

// ---------------- output offsets (floats) ----------------
#define ATTN_OFF  ((size_t)BATCH * MAX_T * VOCAB)                   // 131072000
#define H_OFF     (ATTN_OFF + (size_t)BATCH * MAX_T * T_ENC)       // 133169152
#define C_OFF     (H_OFF + (size_t)BATCH * DHID)                   // 133201920

// ---------------- workspace offsets (bytes) ----------------
#define WS_ENCB   0ULL                               // bf16 enc      33554432 B
#define WS_WFC    33554432ULL                        // bf16           2097152 B
#define WS_WCAT   35651584ULL                        // bf16          20971520 B
#define WS_WWORD  56623104ULL                        // bf16           2621440 B
#define WS_WEMB   59244544ULL                        // bf16          32768000 B
#define WS_BGATES 92012544ULL                        // f32              16384 B
#define WS_ASWX   92028928ULL                        // bf16 A [we|e|h] 163840 B
#define WS_ASWY   92192768ULL                        // bf16 A [h|we|e] 163840 B
#define WS_ASWW   92356608ULL                        // bf16 A word_emb  32768 B
#define WS_E      92389376ULL                        // f32 32x512       65536 B
#define WS_Q      92454912ULL                        // f32 32x1024     131072 B
#define WS_GATES  92585984ULL                        // f32 32x4096     524288 B
#define WS_WE32   93110272ULL                        // f32 32x512       65536 B
#define WS_C      93175808ULL                        // f32 32x1024     131072 B

typedef __bf16 bf16x16 __attribute__((ext_vector_type(16)));
typedef float  fx8     __attribute__((ext_vector_type(8)));

// Scatter one f32 value into the fragment-swizzled bf16 A buffer.
// Layout: A[((mTile*aKS + ks)*32 + lane)*16 + e]
// ISA 16-bit A 16x32 mapping inverse: kl=k%32; half=(kl>>3)&1; e=(kl&7)+(kl>=16?8:0)
__device__ __forceinline__ void write_afrag(__bf16* __restrict__ A, int aKS,
                                            int row, int k, float v) {
    int mt = row >> 4, r = row & 15;
    int ks = k >> 5, kl = k & 31;
    int half = (kl >> 3) & 1;
    int e = (kl & 7) + ((kl >> 4) << 3);
    int lane = r + (half << 4);
    A[(((size_t)mt * aKS + ks) * 32 + lane) * 16 + e] = (__bf16)v;
}

// =====================================================================
// Setup kernels
// =====================================================================
__global__ void k_cvt_bf16(const float* __restrict__ src, __bf16* __restrict__ dst, size_t n) {
    size_t i = (size_t)blockIdx.x * blockDim.x + threadIdx.x;
    if (i < n) dst[i] = (__bf16)src[i];
}

// Swizzle row-major (N x srcK) f32 weight into WMMA B-fragment order:
// dst[((nTile*dstKS + ks)*32 + lane)*16 + e], B[k][n]=W[n][k],
// lane=(n&15)+16*((k%32>>3)&1), e=(k&7)+8*(k%32>=16)  (ISA 16-bit B 32x16 layout)
__global__ void k_swizzleB(const float* __restrict__ src, __bf16* __restrict__ dst,
                           int N, int srcK, int dstKS, int ksOff) {
    size_t tid = (size_t)blockIdx.x * blockDim.x + threadIdx.x;
    size_t total = (size_t)N * srcK;
    if (tid >= total) return;
    int n = (int)(tid / srcK);
    int k = (int)(tid % srcK);
    int kd   = ksOff * 32 + k;
    int ks   = kd >> 5;
    int kl   = kd & 31;
    int half = (kl >> 3) & 1;
    int e    = (kl & 7) + ((kl >> 4) << 3);
    int lane = (n & 15) + (half << 4);
    int nt   = n >> 4;
    dst[(((size_t)nt * dstKS + ks) * 32 + lane) * 16 + e] = (__bf16)src[tid];
}

__global__ void k_bias_add(const float* a, const float* b, float* c, int n) {
    int i = blockIdx.x * blockDim.x + threadIdx.x;
    if (i < n) c[i] = a[i] + b[i];
}

__global__ void k_init_state(const float* __restrict__ h0, const float* __restrict__ c0,
                             const float* __restrict__ go,
                             __bf16* __restrict__ AswX, __bf16* __restrict__ AswY,
                             float* __restrict__ E, float* __restrict__ C) {
    int i = blockIdx.x * blockDim.x + threadIdx.x;          // 32*1024 threads
    if (i >= BATCH * DHID) return;
    int b = i / DHID, d = i % DHID;
    float h = h0[i];
    write_afrag(AswY, KCONT / 32, b, d, h);                 // h slot of cont
    write_afrag(AswX, KCAT / 32, b, DHID + DEMB + d, h);    // h slot of gate input
    C[i] = c0[i];
    if (d < DEMB) E[b * DEMB + d] = go[d];                  // e_0 = go broadcast
}

// =====================================================================
// WMMA GEMM: D(32 x N) = Asw(bf16, fragment order) * Bsw(bf16) + bias
// One wave/block computes the FULL 32-row M extent x four 16-col N tiles
// (32x64 per wave, 8 WMMAs per k-step) so B fragments are loaded once.
// k-loop is software-pipelined: next k-step fragments are loaded before
// the current WMMAs are issued.
// Optional aux output scatters D (incl. bias) into a swizzled-A buffer.
// =====================================================================
__global__ __launch_bounds__(32)
void k_gemm_wmma(const __bf16* __restrict__ Asw, int aKS,
                 const __bf16* __restrict__ B,
                 const float* __restrict__ bias,
                 float* __restrict__ D, unsigned long long dBase, long long ldd,
                 int KS, __bf16* __restrict__ aux, int auxKS) {
    const int lane = threadIdx.x;
    const int half = lane >> 4;
    const int n0   = blockIdx.x << 6;          // 64 columns per wave

    const __bf16* a0p = Asw + (size_t)lane * 16;                      // M tile 0
    const __bf16* a1p = Asw + ((size_t)aKS * 32 + lane) * 16;         // M tile 1
    const __bf16* bp  = B + ((size_t)(n0 >> 4) * KS * 32 + lane) * 16;
    const size_t bT = (size_t)KS * 512;        // elements per nTile

    fx8 z = {0.f,0.f,0.f,0.f,0.f,0.f,0.f,0.f};
    fx8 c00 = z, c01 = z, c02 = z, c03 = z;    // M tile 0 x N tiles 0..3
    fx8 c10 = z, c11 = z, c12 = z, c13 = z;    // M tile 1 x N tiles 0..3

    // prologue: load k-step 0
    bf16x16 a0 = *(const bf16x16*)(a0p);
    bf16x16 a1 = *(const bf16x16*)(a1p);
    bf16x16 b0 = *(const bf16x16*)(bp);
    bf16x16 b1 = *(const bf16x16*)(bp + bT);
    bf16x16 b2 = *(const bf16x16*)(bp + 2 * bT);
    bf16x16 b3 = *(const bf16x16*)(bp + 3 * bT);

    for (int ks = 0; ks < KS; ++ks) {
        // issue next k-step loads first (clamped; last iter reloads harmlessly)
        const int kn = (ks + 1 < KS) ? (ks + 1) : ks;
        const size_t o = (size_t)kn * 512;
        bf16x16 a0n = *(const bf16x16*)(a0p + o);
        bf16x16 a1n = *(const bf16x16*)(a1p + o);
        bf16x16 b0n = *(const bf16x16*)(bp + o);
        bf16x16 b1n = *(const bf16x16*)(bp + bT + o);
        bf16x16 b2n = *(const bf16x16*)(bp + 2 * bT + o);
        bf16x16 b3n = *(const bf16x16*)(bp + 3 * bT + o);

        c00 = __builtin_amdgcn_wmma_f32_16x16x32_bf16(false, a0, false, b0, (short)0, c00, false, false);
        c01 = __builtin_amdgcn_wmma_f32_16x16x32_bf16(false, a0, false, b1, (short)0, c01, false, false);
        c02 = __builtin_amdgcn_wmma_f32_16x16x32_bf16(false, a0, false, b2, (short)0, c02, false, false);
        c03 = __builtin_amdgcn_wmma_f32_16x16x32_bf16(false, a0, false, b3, (short)0, c03, false, false);
        c10 = __builtin_amdgcn_wmma_f32_16x16x32_bf16(false, a1, false, b0, (short)0, c10, false, false);
        c11 = __builtin_amdgcn_wmma_f32_16x16x32_bf16(false, a1, false, b1, (short)0, c11, false, false);
        c12 = __builtin_amdgcn_wmma_f32_16x16x32_bf16(false, a1, false, b2, (short)0, c12, false, false);
        c13 = __builtin_amdgcn_wmma_f32_16x16x32_bf16(false, a1, false, b3, (short)0, c13, false, false);

        a0 = a0n; a1 = a1n; b0 = b0n; b1 = b1n; b2 = b2n; b3 = b3n;
    }

    const int col = lane & 15;
    float bia0 = 0.f, bia1 = 0.f, bia2 = 0.f, bia3 = 0.f;
    if (bias) {
        bia0 = bias[n0 + col];
        bia1 = bias[n0 + 16 + col];
        bia2 = bias[n0 + 32 + col];
        bia3 = bias[n0 + 48 + col];
    }
    #pragma unroll
    for (int j = 0; j < 8; ++j) {
        const int rloc = j + 8 * half;
        // M tile 0 (rows 0..15)
        {
            const size_t row = (size_t)rloc;
            float v0 = c00[j] + bia0, v1 = c01[j] + bia1;
            float v2 = c02[j] + bia2, v3 = c03[j] + bia3;
            D[dBase + row * (size_t)ldd + n0 + col]      = v0;
            D[dBase + row * (size_t)ldd + n0 + 16 + col] = v1;
            D[dBase + row * (size_t)ldd + n0 + 32 + col] = v2;
            D[dBase + row * (size_t)ldd + n0 + 48 + col] = v3;
            if (aux) {
                write_afrag(aux, auxKS, (int)row, n0 + col,      v0);
                write_afrag(aux, auxKS, (int)row, n0 + 16 + col, v1);
                write_afrag(aux, auxKS, (int)row, n0 + 32 + col, v2);
                write_afrag(aux, auxKS, (int)row, n0 + 48 + col, v3);
            }
        }
        // M tile 1 (rows 16..31)
        {
            const size_t row = (size_t)(16 + rloc);
            float v0 = c10[j] + bia0, v1 = c11[j] + bia1;
            float v2 = c12[j] + bia2, v3 = c13[j] + bia3;
            D[dBase + row * (size_t)ldd + n0 + col]      = v0;
            D[dBase + row * (size_t)ldd + n0 + 16 + col] = v1;
            D[dBase + row * (size_t)ldd + n0 + 32 + col] = v2;
            D[dBase + row * (size_t)ldd + n0 + 48 + col] = v3;
            if (aux) {
                write_afrag(aux, auxKS, (int)row, n0 + col,      v0);
                write_afrag(aux, auxKS, (int)row, n0 + 16 + col, v1);
                write_afrag(aux, auxKS, (int)row, n0 + 32 + col, v2);
                write_afrag(aux, auxKS, (int)row, n0 + 48 + col, v3);
            }
        }
    }
}

// =====================================================================
// Attention: raw scores -> softmax -> weighted sum, fused per batch b.
// Writes score straight into attn output; writes weighted & e_t as
// swizzled A fragments for the two downstream GEMMs.
// =====================================================================
__global__ __launch_bounds__(512)
void k_attn(const __bf16* __restrict__ encb, const float* __restrict__ q,
            const float* __restrict__ E,
            __bf16* __restrict__ AswX, __bf16* __restrict__ AswY,
            float* __restrict__ attn_out, int t) {
    __shared__ float sq[DHID];
    __shared__ float sc[T_ENC];
    __shared__ float red[16];

    const int b    = blockIdx.x;
    const int tid  = threadIdx.x;
    const int wave = tid >> 5;
    const int lane = tid & 31;

    sq[tid]       = q[b * DHID + tid];
    sq[tid + 512] = q[b * DHID + tid + 512];
    __syncthreads();

    const __bf16* encB = encb + (size_t)b * T_ENC * DHID;

    // raw[t] = dot(enc[b,t,:], q[b,:]) ; wave per row, lanes split K
    for (int tr = wave; tr < T_ENC; tr += 16) {
        const __bf16* er = encB + (size_t)tr * DHID + lane * 32;
        const float*  qp = sq + lane * 32;
        float s = 0.f;
        #pragma unroll
        for (int i = 0; i < 32; ++i) s += (float)er[i] * qp[i];
        #pragma unroll
        for (int off = 16; off >= 1; off >>= 1) s += __shfl_xor(s, off, 32);
        if (lane == 0) sc[tr] = s;
    }
    __syncthreads();

    // softmax over 512 scores
    float v = sc[tid];
    float m = v;
    #pragma unroll
    for (int off = 16; off >= 1; off >>= 1) m = fmaxf(m, __shfl_xor(m, off, 32));
    if (lane == 0) red[wave] = m;
    __syncthreads();
    float M = red[0];
    #pragma unroll
    for (int i = 1; i < 16; ++i) M = fmaxf(M, red[i]);
    __syncthreads();
    float ex = __expf(v - M);
    float s2 = ex;
    #pragma unroll
    for (int off = 16; off >= 1; off >>= 1) s2 += __shfl_xor(s2, off, 32);
    if (lane == 0) red[wave] = s2;
    __syncthreads();
    float S = 0.f;
    #pragma unroll
    for (int i = 0; i < 16; ++i) S += red[i];
    float p = ex / S;
    sc[tid] = p;
    attn_out[((size_t)b * MAX_T + t) * T_ENC + tid] = p;
    __syncthreads();

    // weighted[d] = sum_t score[t] * enc[b,t,d]  (coalesced over d)
    for (int d = tid; d < DHID; d += 512) {
        const __bf16* ec = encB + d;
        float acc = 0.f;
        #pragma unroll 8
        for (int tt = 0; tt < T_ENC; ++tt) acc += sc[tt] * (float)ec[(size_t)tt * DHID];
        write_afrag(AswX, KCAT / 32, b, d, acc);           // [weighted | e | h]
        write_afrag(AswY, KCONT / 32, b, DHID + d, acc);   // [h_t | weighted | e]
    }
    // e_t into both concat A buffers
    float ev = E[b * DEMB + tid];
    write_afrag(AswX, KCAT / 32, b, DHID + tid, ev);
    write_afrag(AswY, KCONT / 32, b, DHID + DHID + tid, ev);
}

// =====================================================================
// LSTM activations + teacher-forcing embedding gather
// =====================================================================
__global__ __launch_bounds__(256)
void k_lstm(const float* __restrict__ gates, float* __restrict__ C,
            __bf16* __restrict__ AswX, __bf16* __restrict__ AswY,
            float* __restrict__ E,
            const float* __restrict__ embf, const int* __restrict__ gold, int t,
            float* __restrict__ outH, float* __restrict__ outC, int writeHC) {
    const int b = blockIdx.x, tid = threadIdx.x;
    for (int d = tid; d < DHID; d += 256) {
        float gi  = gates[b * NGATES + d];
        float gf  = gates[b * NGATES + DHID + d];
        float gg  = gates[b * NGATES + 2 * DHID + d];
        float go_ = gates[b * NGATES + 3 * DHID + d];
        float si = 1.f / (1.f + __expf(-gi));
        float sf = 1.f / (1.f + __expf(-gf));
        float so = 1.f / (1.f + __expf(-go_));
        float cn = sf * C[b * DHID + d] + si * tanhf(gg);
        float hn = so * tanhf(cn);
        C[b * DHID + d] = cn;
        write_afrag(AswY, KCONT / 32, b, d, hn);                 // cont h / next-q h
        write_afrag(AswX, KCAT / 32, b, DHID + DEMB + d, hn);    // next gate-input h
        if (writeHC) { outH[b * DHID + d] = hn; outC[b * DHID + d] = cn; }
    }
    const int w = gold[b * MAX_T + t];
    for (int d = tid; d < DEMB; d += 256)
        E[b * DEMB + d] = embf[(size_t)w * DEMB + d];
}

// =====================================================================
// host-side launcher
// =====================================================================
extern "C" void kernel_launch(void* const* d_in, const int* in_sizes, int n_in,
                              void* d_out, int out_size, void* d_ws, size_t ws_size,
                              hipStream_t stream) {
    (void)in_sizes; (void)n_in; (void)out_size; (void)ws_size;
    const float* enc  = (const float*)d_in[0];
    const int*   gold = (const int*)  d_in[1];
    const float* h0   = (const float*)d_in[2];
    const float* c0   = (const float*)d_in[3];
    const float* embw = (const float*)d_in[4];
    const float* go   = (const float*)d_in[5];
    const float* Wfc  = (const float*)d_in[6];
    const float* bfc  = (const float*)d_in[7];
    const float* Wih  = (const float*)d_in[8];
    const float* bih  = (const float*)d_in[9];
    const float* Whh  = (const float*)d_in[10];
    const float* bhh  = (const float*)d_in[11];
    const float* Www  = (const float*)d_in[12];
    const float* bww  = (const float*)d_in[13];

    char* w = (char*)d_ws;
    __bf16* encB  = (__bf16*)(w + WS_ENCB);
    __bf16* WfcS  = (__bf16*)(w + WS_WFC);
    __bf16* WcatS = (__bf16*)(w + WS_WCAT);
    __bf16* WwdS  = (__bf16*)(w + WS_WWORD);
    __bf16* WemS  = (__bf16*)(w + WS_WEMB);
    __bf16* AswX  = (__bf16*)(w + WS_ASWX);
    __bf16* AswY  = (__bf16*)(w + WS_ASWY);
    __bf16* AswW  = (__bf16*)(w + WS_ASWW);
    float* bg  = (float*)(w + WS_BGATES);
    float* E   = (float*)(w + WS_E);
    float* Q   = (float*)(w + WS_Q);
    float* G   = (float*)(w + WS_GATES);
    float* WE  = (float*)(w + WS_WE32);
    float* C   = (float*)(w + WS_C);

    float* out  = (float*)d_out;
    float* attn = out + ATTN_OFF;
    float* outH = out + H_OFF;
    float* outC = out + C_OFF;

    // ---- one-time setup (bf16 conversion + fragment swizzle) ----
    {
        size_t nEnc = (size_t)BATCH * T_ENC * DHID;
        k_cvt_bf16<<<(int)((nEnc + 255) / 256), 256, 0, stream>>>(enc, encB, nEnc);
        size_t n;
        n = (size_t)DHID * DHID;
        k_swizzleB<<<(int)((n + 255) / 256), 256, 0, stream>>>(Wfc, WfcS, DHID, DHID, DHID / 32, 0);
        n = (size_t)NGATES * (DHID + DEMB);
        k_swizzleB<<<(int)((n + 255) / 256), 256, 0, stream>>>(Wih, WcatS, NGATES, DHID + DEMB, KCAT / 32, 0);
        n = (size_t)NGATES * DHID;
        k_swizzleB<<<(int)((n + 255) / 256), 256, 0, stream>>>(Whh, WcatS, NGATES, DHID, KCAT / 32, (DHID + DEMB) / 32);
        n = (size_t)DEMB * KCONT;
        k_swizzleB<<<(int)((n + 255) / 256), 256, 0, stream>>>(Www, WwdS, DEMB, KCONT, KCONT / 32, 0);
        n = (size_t)VOCAB * DEMB;
        k_swizzleB<<<(int)((n + 255) / 256), 256, 0, stream>>>(embw, WemS, VOCAB, DEMB, DEMB / 32, 0);
        k_bias_add<<<NGATES / 256, 256, 0, stream>>>(bih, bhh, bg, NGATES);
        k_init_state<<<BATCH * DHID / 256, 256, 0, stream>>>(h0, c0, go, AswX, AswY, E, C);
    }

    // ---- 128 sequential decode steps ----
    for (int t = 0; t < MAX_T; ++t) {
        // q = h @ W_fc^T + b_fc   (h = leading ks-slots of AswY)
        k_gemm_wmma<<<DHID / 64, 32, 0, stream>>>(
            AswY, KCONT / 32, WfcS, bfc, Q, 0ULL, DHID, DHID / 32, nullptr, 0);
        // attention (raw -> softmax -> weighted) + e_t staging
        k_attn<<<BATCH, 512, 0, stream>>>(encB, Q, E, AswX, AswY, attn, t);
        // gates = [weighted|e|h] @ [W_ih|W_hh]^T + (b_ih+b_hh)
        k_gemm_wmma<<<NGATES / 64, 32, 0, stream>>>(
            AswX, KCAT / 32, WcatS, bg, G, 0ULL, NGATES, KCAT / 32, nullptr, 0);
        // LSTM cell + teacher-forcing gather
        k_lstm<<<BATCH, 256, 0, stream>>>(G, C, AswX, AswY, E, embw, gold, t,
                                          outH, outC, (t == MAX_T - 1) ? 1 : 0);
        // word_emb = [h_t|weighted|e] @ W_word^T + b_word ; aux -> swizzled A for logits
        k_gemm_wmma<<<DEMB / 64, 32, 0, stream>>>(
            AswY, KCONT / 32, WwdS, bww, WE, 0ULL, DEMB, KCONT / 32, AswW, DEMB / 32);
        // logits = word_emb @ emb^T  -> written straight into words[b,t,:]
        k_gemm_wmma<<<VOCAB / 64, 32, 0, stream>>>(
            AswW, DEMB / 32, WemS, nullptr, out,
            (unsigned long long)t * VOCAB, (long long)MAX_T * VOCAB, DEMB / 32, nullptr, 0);
    }
}
